// Seq2Seq_69690139345320
// MI455X (gfx1250) — matrix operand order repaired and checked by reading, not compiled, not run on hardware
//
#include <hip/hip_runtime.h>
#include <hip/hip_bf16.h>
#include <cstddef>

typedef __attribute__((ext_vector_type(16))) _Float16 v16h;
typedef __attribute__((ext_vector_type(8)))  float    v8f;

constexpr int BTOT = 4096;   // batch
constexpr int SENC = 512;    // encoder seq
constexpr int SP   = 513;    // padded encoder seq
constexpr int HID  = 16;     // hidden
constexpr int NV   = 6;      // classes
constexpr int TD   = 64;     // decoder steps (T+1)
constexpr int FDIM = 14;     // cont features

__device__ __forceinline__ v8f wmma16(v16h a, v16h b, v8f c) {
  // v_wmma_f32_16x16x32_f16: D = A(16x32 f16) x B(32x16 f16) + C(16x16 f32)
  return __builtin_amdgcn_wmma_f32_16x16x32_f16(false, a, false, b, (short)0, c,
                                                false, false);
}
// Fast transcendental helpers: single v_exp_f32 / v_rcp_f32 trans ops (no IEEE
// div refinement chain) -- these sit on the serial GRU recurrence, so latency
// matters more than the last ulp next to f16 WMMA inputs.
__device__ __forceinline__ float rcp_(float x) { return __builtin_amdgcn_rcpf(x); }
__device__ __forceinline__ float sigm_(float x) { return rcp_(1.0f + __expf(-x)); }
__device__ __forceinline__ float tanh_(float x) {
  return 1.0f - 2.0f * rcp_(__expf(2.0f * x) + 1.0f);
}

// ---------------------------------------------------------------------------
// Encoder: one wave (32 threads) per 16 batch rows; 513 sequential GRU steps.
// A-layout (f16 16x32): lane<16 holds row m=lane, K 0..7 (v[0..7]) and 16..23
// (v[8..15]); lane>=16 holds row m=lane-16, K 8..15 and 24..31.
// B-layout (f16 32x16): lane&15 = output column n; lanes<16 hold K=0..7 and 16..23
// packed pairwise, lanes>=16 hold K=8..15 and 24..31.
// C-layout (f32 16x16): element g of v8f = row m = g + 8*(lane>=16), col n=lane&15.
// ---------------------------------------------------------------------------
__global__ __launch_bounds__(32) void s2s_encoder(
    const float* __restrict__ cont, const int* __restrict__ words,
    const float* __restrict__ Ew,
    const float* __restrict__ Wih, const float* __restrict__ Whh,
    const float* __restrict__ bih, const float* __restrict__ bhh,
    float* __restrict__ M, float* __restrict__ hT) {
  __shared__ float sEw[NV * HID];
  __shared__ float sH[16 * 16];

  const int lane = threadIdx.x;
  const int b0 = blockIdx.x * 16;
  const int n = lane & 15;             // WMMA column / hidden index
  const int m = lane & 15;             // batch row within tile (A layout)
  const int hi = (lane >> 4) & 1;      // lane half
  const int kb = hi ? 16 : 0;

  for (int i = lane; i < NV * HID; i += 32) sEw[i] = Ew[i];
  for (int i = lane; i < 256; i += 32) sH[i] = 0.0f;

  // Pre-pack gate weights into f16 WMMA B-layout registers (stay resident).
  v16h Bi[3], Bh[3];
  float biv[3], bhv[3];
#pragma unroll
  for (int c = 0; c < 3; ++c) {
    const int row = c * 16 + n;        // gate unit (0..47)
#pragma unroll
    for (int e = 0; e < 16; ++e) Bi[c][e] = (_Float16)Wih[row * 32 + 8 * hi + e + (e >= 8 ? 8 : 0)];
#pragma unroll
    for (int e = 0; e < 16; ++e)
      Bh[c][e] = (e >= 8) ? (_Float16)0.0f : (_Float16)Whh[row * 16 + 8 * hi + e];  // K pad 16->32
    biv[c] = bih[row];
    bhv[c] = bhh[row];
  }

  v8f hC = {0.f, 0.f, 0.f, 0.f, 0.f, 0.f, 0.f, 0.f};  // h in C layout
  __syncthreads();

  for (int t = 0; t < SP; ++t) {
    const int w = (t < SENC) ? words[(size_t)(b0 + m) * SENC + t] : 2 /*WEOS*/;
    const float validf = (w != 0) ? 1.0f : 0.0f;       // mask = (word == PAD)

    // ---- build x_t A tile: dims 0..15 = Ew[w], 16 = SOS, 17..30 = cont, 31 = EOS
    v16h xA;
    const float* ep = &sEw[w * HID + (hi ? 8 : 0)];
#pragma unroll
    for (int e = 0; e < 8; ++e) xA[e] = (_Float16)ep[e];
    if (t < SENC) {
      const float* cp = &cont[((size_t)(b0 + m) * SENC + t) * FDIM];
      if (!hi) {
        xA[8] = (_Float16)1.0f;                         // SOS pad column
#pragma unroll
        for (int e = 0; e < 7; ++e) xA[9 + e] = (_Float16)cp[e];
      } else {
#pragma unroll
        for (int e = 0; e < 7; ++e) xA[8 + e] = (_Float16)cp[7 + e];
        xA[15] = (_Float16)2.0f;                        // EOS pad column
      }
    } else {
#pragma unroll
      for (int e = 8; e < 16; ++e) xA[e] = (_Float16)0.0f;  // PAD time row
    }

    // ---- h in A layout (K 0..15 = h, 16..31 = 0), via LDS transpose
    v16h hA;
    const float* hp = &sH[m * 16 + (hi ? 8 : 0)];
#pragma unroll
    for (int e = 0; e < 8; ++e) { hA[e] = (_Float16)hp[e]; hA[8 + e] = (_Float16)0.0f; }

    const v8f z8 = {0.f, 0.f, 0.f, 0.f, 0.f, 0.f, 0.f, 0.f};
    v8f gi0 = wmma16(xA, Bi[0], z8);
    v8f gi1 = wmma16(xA, Bi[1], z8);
    v8f gi2 = wmma16(xA, Bi[2], z8);
    v8f gh0 = wmma16(hA, Bh[0], z8);
    v8f gh1 = wmma16(hA, Bh[1], z8);
    v8f gh2 = wmma16(hA, Bh[2], z8);

#pragma unroll
    for (int g = 0; g < 8; ++g) {
      const float r  = sigm_((gi0[g] + biv[0]) + (gh0[g] + bhv[0]));
      const float z  = sigm_((gi1[g] + biv[1]) + (gh1[g] + bhv[1]));
      const float nn = tanh_((gi2[g] + biv[2]) + r * (gh2[g] + bhv[2]));
      const float hn = (1.0f - z) * nn + z * hC[g];
      const int mm = g + (hi << 3);                     // C-layout row
      const float vs = __shfl(validf, mm, 32);          // valid flag of row mm
      const float hsel = (vs > 0.5f) ? hn : hC[g];
      const float mout = (vs > 0.5f) ? hn : 0.0f;       // pad_packed zeros
      hC[g] = hsel;
      M[((size_t)(b0 + mm) * SP + t) * HID + n] = mout;
      sH[mm * 16 + n] = hsel;                           // for next step's hA
    }
  }
#pragma unroll
  for (int g = 0; g < 8; ++g) {
    const int mm = g + (hi << 3);
    hT[(size_t)(b0 + mm) * HID + n] = hC[g];
  }
}

// ---------------------------------------------------------------------------
// Decoder: 256 threads (8 waves) per 16 batch rows. Wave0 owns the WMMA GRU /
// projections; all waves cooperate on attention over 513 positions (M is
// L2-resident: 134 MB < 192 MB L2). 64 barrier-phased sequential steps.
// ---------------------------------------------------------------------------
__global__ __launch_bounds__(256) void s2s_decoder(
    const int* __restrict__ targ, const float* __restrict__ Ed,
    const float* __restrict__ Wih, const float* __restrict__ Whh,
    const float* __restrict__ bih, const float* __restrict__ bhh,
    const float* __restrict__ Wc, const float* __restrict__ bc,
    const float* __restrict__ Wout, const float* __restrict__ bout,
    const int* __restrict__ words, const float* __restrict__ M,
    const float* __restrict__ hT, float* __restrict__ results,
    float* __restrict__ lossNum, float* __restrict__ lossCnt) {
  __shared__ float sE[SP * 16];         // exp(score)[s][m]
  __shared__ float sH[256], sVa[256], sC[256], sLg[256];
  __shared__ float sEd[NV * HID];
  __shared__ float sSum[16];
  __shared__ int sYi[16];

  const int tid = threadIdx.x;
  const int b0 = blockIdx.x * 16;

  if (tid < NV * HID) sEd[tid] = Ed[tid];
  sH[tid] = hT[(size_t)b0 * HID + tid];                 // h0 = h_enc  [m][j]
  sVa[tid] = 0.0f;                                      // Va0 = 0
  if (tid < 16) sYi[tid] = 1;                           // yi0 = SOS

  // wave0 persistent WMMA state
  v16h Bi[3], Bh[3], BWc, BWo;
  float biv[3] = {0.f, 0.f, 0.f}, bhv[3] = {0.f, 0.f, 0.f};
  float bcn = 0.0f, bon = 0.0f;
  v8f hC = {0.f, 0.f, 0.f, 0.f, 0.f, 0.f, 0.f, 0.f};
  const int lane = tid;
  const int n = lane & 15;
  const int hi = (lane >> 4) & 1;
  if (tid < 32) {
    const int kb = hi ? 16 : 0;
#pragma unroll
    for (int c = 0; c < 3; ++c) {
      const int row = c * 16 + n;
#pragma unroll
      for (int e = 0; e < 16; ++e) Bi[c][e] = (_Float16)Wih[row * 32 + 8 * hi + e + (e >= 8 ? 8 : 0)];
#pragma unroll
      for (int e = 0; e < 16; ++e)
        Bh[c][e] = (e >= 8) ? (_Float16)0.0f : (_Float16)Whh[row * 16 + 8 * hi + e];
      biv[c] = bih[row];
      bhv[c] = bhh[row];
    }
#pragma unroll
    for (int e = 0; e < 16; ++e) BWc[e] = (_Float16)Wc[n * 32 + 8 * hi + e + (e >= 8 ? 8 : 0)];
#pragma unroll
    for (int e = 0; e < 16; ++e)
      BWo[e] = (e >= 8 || n >= NV) ? (_Float16)0.0f : (_Float16)Wout[n * HID + 8 * hi + e];
    bcn = bc[n];
    bon = (n < NV) ? bout[n] : 0.0f;
  }
  __syncthreads();
  if (tid < 32) {
#pragma unroll
    for (int g = 0; g < 8; ++g) hC[g] = sH[(g + (hi << 3)) * 16 + n];
  }

  const v8f z8 = {0.f, 0.f, 0.f, 0.f, 0.f, 0.f, 0.f, 0.f};

  for (int t = 0; t < TD; ++t) {
    // ---- phase 1: GRU step on wave0 (x = [Ed[yi] || Va], 16x32)
    if (tid < 32) {
      const int m = n;
      const int yi = sYi[m];
      v16h xA, hA;
      const float* ep = &sEd[yi * HID + (hi ? 8 : 0)];
      const float* vp = &sVa[m * 16 + (hi ? 8 : 0)];
      const float* hp = &sH[m * 16 + (hi ? 8 : 0)];
#pragma unroll
      for (int e = 0; e < 8; ++e) {
        xA[e] = (_Float16)ep[e];
        xA[8 + e] = (_Float16)vp[e];
        hA[e] = (_Float16)hp[e];
        hA[8 + e] = (_Float16)0.0f;
      }
      v8f gi0 = wmma16(xA, Bi[0], z8);
      v8f gi1 = wmma16(xA, Bi[1], z8);
      v8f gi2 = wmma16(xA, Bi[2], z8);
      v8f gh0 = wmma16(hA, Bh[0], z8);
      v8f gh1 = wmma16(hA, Bh[1], z8);
      v8f gh2 = wmma16(hA, Bh[2], z8);
#pragma unroll
      for (int g = 0; g < 8; ++g) {
        const float r  = sigm_((gi0[g] + biv[0]) + (gh0[g] + bhv[0]));
        const float z  = sigm_((gi1[g] + biv[1]) + (gh1[g] + bhv[1]));
        const float nn = tanh_((gi2[g] + biv[2]) + r * (gh2[g] + bhv[2]));
        const float hn = (1.0f - z) * nn + z * hC[g];
        hC[g] = hn;
        sH[(g + (hi << 3)) * 16 + n] = hn;
      }
    }
    if (tid >= 32 && tid < 48) sSum[tid - 32] = 0.0f;
    __syncthreads();

    // ---- phase 2: scores[m][s] = h[m].M[b,s,:]; e = exp(score) (|score|<=16)
    {
      const int m = tid >> 4;
      const int sb = tid & 15;
      float hreg[16];
#pragma unroll
      for (int j = 0; j < 16; ++j) hreg[j] = sH[m * 16 + j];
      float lsum = 0.0f;
      const int* wrow = &words[(size_t)(b0 + m) * SENC];
      const float* Mrow = &M[(size_t)(b0 + m) * SP * HID];
      for (int s = sb; s < SP; s += 16) {
        const float* Mp = Mrow + (size_t)s * HID;
        __builtin_prefetch(Mp + 16 * HID, 0, 1);        // global_prefetch next stripe
        float sc = 0.0f;
#pragma unroll
        for (int j = 0; j < 16; ++j) sc += hreg[j] * Mp[j];
        const int w = (s < SENC) ? wrow[s] : 2;
        const float e = (w == 0) ? 0.0f : __expf(sc);   // masked -> weight 0
        sE[s * 16 + m] = e;
        lsum += e;
      }
      atomicAdd(&sSum[m], lsum);                        // ds_add_f32
    }
    __syncthreads();

    // ---- phase 3: context c[m][j] = sum_s e * M / sum
    {
      const int m = tid >> 4;
      const int j = tid & 15;
      const float* Mp = &M[(size_t)(b0 + m) * SP * HID + j];
      float acc = 0.0f;
      for (int s = 0; s < SP; ++s) acc += sE[s * 16 + m] * Mp[(size_t)s * HID];
      sC[m * 16 + j] = acc * rcp_(sSum[m]);             // v_rcp_f32, no div chain
    }
    __syncthreads();

    // ---- phase 4: Va = tanh([c||h]Wc^T+bc); logits = Va Wout^T + bout; NLL
    if (tid < 32) {
      const int m = n;
      v16h aV;
      const float* cp = &sC[m * 16 + (hi ? 8 : 0)];
      const float* hp = &sH[m * 16 + (hi ? 8 : 0)];
#pragma unroll
      for (int e = 0; e < 8; ++e) { aV[e] = (_Float16)cp[e]; aV[8 + e] = (_Float16)hp[e]; }
      v8f vaC = wmma16(aV, BWc, z8);
#pragma unroll
      for (int g = 0; g < 8; ++g)
        sVa[(g + (hi << 3)) * 16 + n] = tanh_(vaC[g] + bcn);

      v16h aL;
      const float* vp = &sVa[m * 16 + (hi ? 8 : 0)];
#pragma unroll
      for (int e = 0; e < 8; ++e) { aL[e] = (_Float16)vp[e]; aL[8 + e] = (_Float16)0.0f; }
      v8f lgC = wmma16(aL, BWo, z8);
#pragma unroll
      for (int g = 0; g < 8; ++g)
        sLg[(g + (hi << 3)) * 16 + n] = lgC[g] + bon;

      if (lane < 16) {                                  // one lane per batch row
        const int mm = lane;
        float lg[NV];
        float se = 0.0f;
#pragma unroll
        for (int v = 0; v < NV; ++v) { lg[v] = sLg[mm * 16 + v]; se += __expf(lg[v]); }
        const float lse = __logf(se);
        float* rp = &results[((size_t)t * BTOT + (b0 + mm)) * NV];
#pragma unroll
        for (int v = 0; v < NV; ++v) rp[v] = lg[v] - lse;
        const int lbl = (t < TD - 1) ? targ[(size_t)(b0 + mm) * (TD - 1) + t] : 2;
        const int valid = (lbl != 0);
        atomicAdd(&lossNum[t], valid ? (lg[lbl] - lse) : 0.0f);
        atomicAdd(&lossCnt[t], valid ? 1.0f : 0.0f);
        sYi[mm] = lbl;                                  // next-step input word
      }
    }
    __syncthreads();
  }
}

__global__ void s2s_zero(float* __restrict__ p, int nfl) {
  const int i = blockIdx.x * blockDim.x + threadIdx.x;
  if (i < nfl) p[i] = 0.0f;
}

__global__ void s2s_loss(const float* __restrict__ num, const float* __restrict__ cnt,
                         float* __restrict__ out) {
  if (threadIdx.x == 0 && blockIdx.x == 0) {
    float L = 0.0f;
    for (int t = 0; t < TD; ++t) L += -num[t] / fmaxf(cnt[t], 1.0f);
    out[0] = L;
  }
}

extern "C" void kernel_launch(void* const* d_in, const int* in_sizes, int n_in,
                              void* d_out, int out_size, void* d_ws, size_t ws_size,
                              hipStream_t stream) {
  const float* cont  = (const float*)d_in[0];
  const int*   ewrds = (const int*)d_in[1];
  const int*   targ  = (const int*)d_in[2];
  const float* Ew    = (const float*)d_in[3];
  const float* eWih  = (const float*)d_in[4];
  const float* eWhh  = (const float*)d_in[5];
  const float* ebih  = (const float*)d_in[6];
  const float* ebhh  = (const float*)d_in[7];
  const float* Ed    = (const float*)d_in[8];
  const float* dWih  = (const float*)d_in[9];
  const float* dWhh  = (const float*)d_in[10];
  const float* dbih  = (const float*)d_in[11];
  const float* dbhh  = (const float*)d_in[12];
  const float* Wc    = (const float*)d_in[13];
  const float* bc    = (const float*)d_in[14];
  const float* Wout  = (const float*)d_in[15];
  const float* bout  = (const float*)d_in[16];

  float* out = (float*)d_out;                 // [0] = loss, then [64,4096,6] results
  float* M   = (float*)d_ws;                  // [4096,513,16] f32 (L2-resident)
  float* hT  = M + (size_t)BTOT * SP * HID;   // [4096,16]
  float* lossNum = hT + (size_t)BTOT * HID;   // [64]
  float* lossCnt = lossNum + TD;              // [64]

  s2s_zero<<<1, 128, 0, stream>>>(lossNum, 2 * TD);
  s2s_encoder<<<BTOT / 16, 32, 0, stream>>>(cont, ewrds, Ew, eWih, eWhh, ebih, ebhh,
                                            M, hT);
  s2s_decoder<<<BTOT / 16, 256, 0, stream>>>(targ, Ed, dWih, dWhh, dbih, dbhh, Wc, bc,
                                             Wout, bout, ewrds, M, hT, out + 1,
                                             lossNum, lossCnt);
  s2s_loss<<<1, 1, 0, stream>>>(lossNum, lossCnt, out);
  (void)in_sizes; (void)n_in; (void)out_size; (void)ws_size;
}
